// RnnBptt_48430051229762
// MI455X (gfx1250) — matrix operand order repaired and checked by reading, not compiled, not run on hardware
//
#include <hip/hip_runtime.h>
#include <hip/hip_bf16.h>
#include <math.h>

// RNN scan: T=2048, B=64, D_IN=D_HID=D_OUT=256
#define T_STEPS 2048
#define BATCH   64
#define HID     256
#define BLK_ELEMS (BATCH * HID)   // 16384 f32 per timestep block

typedef __bf16 v16bf __attribute__((ext_vector_type(16)));
typedef float  v8f   __attribute__((ext_vector_type(8)));

union FragU { v16bf v; uint4 q[2]; };

// Split f32 into bf16 hi + bf16 lo residual (truncating; lo captures remainder).
static __device__ __forceinline__ void split_bf16(float v, __bf16& hi, __bf16& lo) {
  unsigned int b = __builtin_bit_cast(unsigned int, v);
  unsigned short h = (unsigned short)(b >> 16);
  float fh = __builtin_bit_cast(float, (unsigned int)((unsigned int)h << 16));
  float r  = v - fh;
  unsigned short l = (unsigned short)(__builtin_bit_cast(unsigned int, r) >> 16);
  hi = __builtin_bit_cast(__bf16, h);
  lo = __builtin_bit_cast(__bf16, l);
}

// A fragment (16x32, 16-bit): M = lane&15, K = k0 + j + 8*(j>=8) + 8*(lane>=16).
// Ab is a row-major [.. , HID] bf16 array -> two contiguous 16B loads per lane.
static __device__ __forceinline__ v16bf load_a_frag(const __bf16* Ab, int m0, int k0, int lane) {
  int m    = m0 + (lane & 15);
  int koff = k0 + ((lane >> 4) << 3);
  const __bf16* p = Ab + m * HID + koff;
  FragU f;
  f.q[0] = *(const uint4*)(p);        // j = 0..7
  f.q[1] = *(const uint4*)(p + 16);   // j = 8..15
  return f.v;
}

// B fragment (32x16, 16-bit): N = lane&15, K = kb*32 + (j&7) + 16*(j>=8) + 8*(lane>=16).
// Bs is fragment-major: [nt][kb][lane][j] contiguous -> 16 contiguous bf16 per lane.
static __device__ __forceinline__ v16bf load_b_frag(const __bf16* Bs, int nt, int kb, int lane) {
  const __bf16* p = Bs + (size_t)((((nt << 3) + kb) << 5) + lane) * 16;
  FragU f;
  f.q[0] = *(const uint4*)(p);
  f.q[1] = *(const uint4*)(p + 8);
  return f.v;
}

#define WMMA_BF16(A, B, C) \
  __builtin_amdgcn_wmma_f32_16x16x32_bf16(false, (A), false, (B), (short)0, (C), false, false)

// ---------------------------------------------------------------------------
// Phase 0: convert W (f32 [n][k], 256x256) into fragment-major split-bf16
// streams: hi[65536] then lo[65536] per matrix. 3 matrices -> 768 KB of d_ws.
// ---------------------------------------------------------------------------
__global__ __launch_bounds__(256) void k_convert(const float* __restrict__ Win,
                                                 const float* __restrict__ Whid,
                                                 const float* __restrict__ Wout,
                                                 __bf16* __restrict__ ws) {
  int idx = blockIdx.x * 256 + threadIdx.x;     // 0 .. 3*65536-1
  int mtx = idx >> 16;
  int e   = idx & 65535;
  const float* W = (mtx == 0) ? Win : ((mtx == 1) ? Whid : Wout);
  __bf16* hi = ws + (size_t)mtx * (2 * 65536);
  __bf16* lo = hi + 65536;
  int j    = e & 15;
  int lane = (e >> 4) & 31;
  int kb   = (e >> 9) & 7;
  int nt   = e >> 12;
  int n = nt * 16 + (lane & 15);
  int k = kb * 32 + (j & 7) + ((j >> 3) << 4) + ((lane >> 4) << 3);
  __bf16 h, l;
  split_bf16(W[n * HID + k], h, l);             // B[k][n] = W[n][k] (transposed)
  hi[e] = h;
  lo[e] = l;
}

// ---------------------------------------------------------------------------
// Phases 1 & 3: out_blk[64x256] = in_blk[64x256] @ Wt (split-bf16 WMMA).
// One block per timestep; safe fully in place (in_blk staged to LDS first).
// 512 threads = 16 waves; each wave: 1 m-tile x 4 n-tiles.
// K-loop unroll capped at 2 to avoid VGPR spills (seen in round-0 asm).
// ---------------------------------------------------------------------------
__global__ __launch_bounds__(512) void k_gemm_blk(const float* __restrict__ in,
                                                  const __bf16* __restrict__ BHi,
                                                  const __bf16* __restrict__ BLo,
                                                  float* __restrict__ out) {
  __shared__ alignas(16) __bf16 Ahi[BLK_ELEMS];
  __shared__ alignas(16) __bf16 Alo[BLK_ELEMS];
  const size_t base = (size_t)blockIdx.x * BLK_ELEMS;
  const float* inb = in + base;
  for (int i = threadIdx.x; i < BLK_ELEMS; i += 512) {
    __bf16 h, l;
    split_bf16(inb[i], h, l);
    Ahi[i] = h;
    Alo[i] = l;
  }
  __syncthreads();

  int wave = threadIdx.x >> 5;
  int lane = threadIdx.x & 31;
  int mt  = wave >> 2;              // 0..3
  int ntg = (wave & 3) << 2;        // n-tile group base (4 tiles)

  v8f zero = {0.f, 0.f, 0.f, 0.f, 0.f, 0.f, 0.f, 0.f};
  v8f acc[4];
  acc[0] = zero; acc[1] = zero; acc[2] = zero; acc[3] = zero;

#pragma clang loop unroll_count(2)
  for (int kb = 0; kb < 8; ++kb) {
    v16bf ahi = load_a_frag(Ahi, mt << 4, kb << 5, lane);
    v16bf alo = load_a_frag(Alo, mt << 4, kb << 5, lane);
#pragma unroll
    for (int q = 0; q < 4; ++q) {
      v16bf bhi = load_b_frag(BHi, ntg + q, kb, lane);
      v16bf blo = load_b_frag(BLo, ntg + q, kb, lane);
      acc[q] = WMMA_BF16(ahi, bhi, acc[q]);
      acc[q] = WMMA_BF16(alo, bhi, acc[q]);
      acc[q] = WMMA_BF16(ahi, blo, acc[q]);
    }
  }

  int mbase = (mt << 4) + ((lane >> 4) << 3);   // C/D: M = r + 8*(lane>=16)
  int nlan  = lane & 15;
  float* outb = out + base;
#pragma unroll
  for (int q = 0; q < 4; ++q) {
    int n = ((ntg + q) << 4) + nlan;
#pragma unroll
    for (int r = 0; r < 8; ++r)
      outb[(mbase + r) * HID + n] = acc[q][r];
  }
}

// ---------------------------------------------------------------------------
// Phase 2: serial scan on ONE WGP. 1024 threads = 32 waves.
// Dynamic LDS (320 KB): W_hidT hi/lo fragment streams (256 KB) + h hi/lo (64 KB).
// buf[t] holds x_proj[t] on input and is overwritten with hs[t] in place.
// ---------------------------------------------------------------------------
__global__ __launch_bounds__(1024) void k_scan(float* __restrict__ buf,
                                               const float* __restrict__ h0,
                                               const __bf16* __restrict__ WHi,
                                               const __bf16* __restrict__ WLo,
                                               float* __restrict__ hlast) {
  extern __shared__ char smem[];
  __bf16* Whi = (__bf16*)smem;          // 65536 elems
  __bf16* Wlo = Whi + 65536;            // 65536
  __bf16* Hhi = Wlo + 65536;            // 16384
  __bf16* Hlo = Hhi + 16384;            // 16384  -> 327680 bytes total

  for (int i = threadIdx.x; i < 65536; i += 1024) {
    Whi[i] = WHi[i];
    Wlo[i] = WLo[i];
  }
  for (int i = threadIdx.x; i < BLK_ELEMS; i += 1024) {
    __bf16 h, l;
    split_bf16(h0[i], h, l);
    Hhi[i] = h;
    Hlo[i] = l;
  }
  __syncthreads();

  int wave = threadIdx.x >> 5;
  int lane = threadIdx.x & 31;
  int mt  = wave >> 3;                  // 0..3
  int nt0 = (wave & 7) << 1;            // even n-tile; wave owns nt0, nt0+1
  int m0  = mt << 4;
  int mbase = m0 + ((lane >> 4) << 3);
  int n0l   = (nt0 << 4) + (lane & 15); // second tile column = n0l + 16 (same 128B line)

  v8f zero = {0.f, 0.f, 0.f, 0.f, 0.f, 0.f, 0.f, 0.f};

  for (int t = 0; t < T_STEPS; ++t) {
    float* blk = buf + (size_t)t * BLK_ELEMS;
    // Prefetch this step's x_proj lines while the WMMA k-loop runs.
#pragma unroll
    for (int r = 0; r < 8; ++r)
      __builtin_prefetch(blk + (mbase + r) * HID + n0l, 0, 3);

    v8f a0 = zero, a1 = zero;
#pragma clang loop unroll_count(2)
    for (int kb = 0; kb < 8; ++kb) {
      v16bf ahi = load_a_frag(Hhi, m0, kb << 5, lane);
      v16bf alo = load_a_frag(Hlo, m0, kb << 5, lane);
      v16bf b0h = load_b_frag(Whi, nt0,     kb, lane);
      v16bf b0l = load_b_frag(Wlo, nt0,     kb, lane);
      v16bf b1h = load_b_frag(Whi, nt0 + 1, kb, lane);
      v16bf b1l = load_b_frag(Wlo, nt0 + 1, kb, lane);
      a0 = WMMA_BF16(ahi, b0h, a0);
      a0 = WMMA_BF16(alo, b0h, a0);
      a0 = WMMA_BF16(ahi, b0l, a0);
      a1 = WMMA_BF16(ahi, b1h, a1);
      a1 = WMMA_BF16(alo, b1h, a1);
      a1 = WMMA_BF16(ahi, b1l, a1);
    }
    __syncthreads();                    // everyone done reading H(t-1)

#pragma unroll
    for (int r = 0; r < 8; ++r) {
      int m  = mbase + r;
      int i0 = m * HID + n0l;
      int i1 = i0 + 16;
      float v0 = tanhf(a0[r] + blk[i0]);
      float v1 = tanhf(a1[r] + blk[i1]);
      blk[i0] = v0;                     // hs[t] overwrites x_proj[t] in place
      blk[i1] = v1;
      __bf16 hh, hl;
      split_bf16(v0, hh, hl); Hhi[i0] = hh; Hlo[i0] = hl;
      split_bf16(v1, hh, hl); Hhi[i1] = hh; Hlo[i1] = hl;
      if (t == T_STEPS - 1) { hlast[i0] = v0; hlast[i1] = v1; }
    }
    __syncthreads();                    // H(t) visible to all waves
  }
}

// ---------------------------------------------------------------------------
extern "C" void kernel_launch(void* const* d_in, const int* in_sizes, int n_in,
                              void* d_out, int out_size, void* d_ws, size_t ws_size,
                              hipStream_t stream) {
  const float* x    = (const float*)d_in[0];   // [T,B,256]
  const float* h0   = (const float*)d_in[1];   // [B,256]
  const float* Win  = (const float*)d_in[2];   // [256,256]
  const float* Whid = (const float*)d_in[3];   // [256,256]
  const float* Wout = (const float*)d_in[4];   // [256,256]
  float* out = (float*)d_out;                  // [T*B,256] ys ++ [B,256] h_last
  __bf16* wsp = (__bf16*)d_ws;                 // needs 768 KB

  const __bf16* WinHi  = wsp;
  const __bf16* WinLo  = wsp + 65536;
  const __bf16* WhidHi = wsp + 131072;
  const __bf16* WhidLo = wsp + 131072 + 65536;
  const __bf16* WoutHi = wsp + 262144;
  const __bf16* WoutLo = wsp + 262144 + 65536;

  // Phase 0: weight transpose + split-bf16 fragment streams (768 blocks).
  k_convert<<<768, 256, 0, stream>>>(Win, Whid, Wout, wsp);

  // Phase 1: x_proj = x @ W_in^T  -> stored in d_out's ys region as scratch.
  k_gemm_blk<<<T_STEPS, 512, 0, stream>>>(x, WinHi, WinLo, out);

  // Phase 2: serial tanh-scan; hs[t] overwrites x_proj[t]; h_last to tail.
  size_t lds_bytes = (size_t)(65536 + 65536 + 16384 + 16384) * sizeof(__bf16); // 320 KB
  k_scan<<<1, 1024, lds_bytes, stream>>>(out, h0, WhidHi, WhidLo,
                                         out + (size_t)T_STEPS * BLK_ELEMS);

  // Phase 3: ys = hs @ W_out^T, fully in place over d_out's ys region.
  k_gemm_blk<<<T_STEPS, 512, 0, stream>>>(out, WoutHi, WoutLo, out);
}
